// GroupedQueryAttention_86973087744170
// MI455X (gfx1250) — matrix operand order repaired and checked by reading, not compiled, not run on hardware
//
#include <hip/hip_runtime.h>

// ---------------------------------------------------------------------------
// GQA attention block for gfx1250 (MI455X), bf16 WMMA everywhere.
// B=2, S=2048, D_IN=4096, D_OUT=4096, H=32, KV=8, HD=128, GROUP=4, causal.
// Async global->LDS double-buffered pipelining via ASYNCcnt.
// ---------------------------------------------------------------------------

typedef __attribute__((ext_vector_type(16))) __bf16 v16bf;
typedef __attribute__((ext_vector_type(8)))  float  v8f;

#if defined(__HIP_DEVICE_COMPILE__) &&                                         \
    __has_builtin(__builtin_amdgcn_global_load_async_to_lds_b128) &&           \
    __has_builtin(__builtin_amdgcn_s_wait_asynccnt)
#define USE_ASYNC 1
#else
#define USE_ASYNC 0
#endif

#if USE_ASYNC
typedef __attribute__((vector_size(16))) int async_v4i;
typedef __attribute__((address_space(1))) async_v4i* async_gptr;
typedef __attribute__((address_space(3))) async_v4i* async_lptr;
#endif

__device__ __forceinline__ void copy16_g2l(__bf16* dst_lds, const __bf16* src_g) {
#if USE_ASYNC
  __builtin_amdgcn_global_load_async_to_lds_b128(
      (async_gptr)src_g, (async_lptr)dst_lds, 0, 0);
#else
  *(uint4*)dst_lds = *(const uint4*)src_g;
#endif
}

// Wait until at most Ncnt async global->LDS ops remain outstanding (per wave).
// Async loads complete in order, so waiting to Ncnt == "drain all but the
// Ncnt just-issued next-tile loads".
template <int Ncnt>
__device__ __forceinline__ void async_wait() {
#if USE_ASYNC
  __builtin_amdgcn_s_wait_asynccnt(Ncnt);
#endif
}

// ---- wave32 WMMA fragment loaders (contiguous k order: lane half h holds k 16h..16h+15) --

__device__ __forceinline__ v16bf load_A_frag(const __bf16* p, int ld) {
  // A: 16x32 bf16. lanes 0-15: M=lane, K 0-15; lanes 16-31: K 16-31.
  const int lane = threadIdx.x & 31;
  const int row  = lane & 15;
  const int kg   = (lane >> 4) << 4;
  v16bf a;
#pragma unroll
  for (int v = 0; v < 8; ++v) {
    const int k = (v << 1) + kg;
    a[2 * v]     = p[row * ld + k];
    a[2 * v + 1] = p[row * ld + k + 1];
  }
  return a;
}

__device__ __forceinline__ v16bf load_B_frag(const __bf16* p, int ld) {
  // B: 32x16 bf16 from a K x N row-major tile (leading dim ld).
  const int lane = threadIdx.x & 31;
  const int n    = lane & 15;
  const int kg   = (lane >> 4) << 4;
  v16bf b;
#pragma unroll
  for (int v = 0; v < 8; ++v) {
    b[2 * v]     = p[(kg + 2 * v) * ld + n];
    b[2 * v + 1] = p[(kg + 2 * v + 1) * ld + n];
  }
  return b;
}

__device__ __forceinline__ v16bf load_B_fragT(const __bf16* p, int ld) {
  // B: 32x16 bf16 from an N x K row-major tile (i.e. B[k][n] = p[n*ld + k]).
  const int lane = threadIdx.x & 31;
  const int n    = lane & 15;
  const int kg   = (lane >> 4) << 4;
  v16bf b;
#pragma unroll
  for (int v = 0; v < 8; ++v) {
    b[2 * v]     = p[n * ld + kg + 2 * v];
    b[2 * v + 1] = p[n * ld + kg + 2 * v + 1];
  }
  return b;
}

__device__ __forceinline__ v8f wmma_bf16(v16bf a, v16bf b, v8f c) {
  return __builtin_amdgcn_wmma_f32_16x16x32_bf16(false, a, false, b,
                                                 (short)0, c, false, false);
}

// ---------------------------- conversion ------------------------------------

__global__ void __launch_bounds__(256)
f32_to_bf16(const float* __restrict__ in, __bf16* __restrict__ out, long n) {
  long i = (long)blockIdx.x * 256 + threadIdx.x;
  if (i < n) out[i] = (__bf16)in[i];
}

// ------------------------ tiled WMMA GEMM (bf16 -> f32) ----------------------
// C[M,N] = A[M,K] * B[K,N]; 128x256 block tile, K-step 32, 8 waves (2x4),
// each wave computes a 64x64 sub-tile = 4x4 WMMA accumulators (16 WMMA/step).
// Double-buffered LDS; async loads for tile i+1 overlap WMMAs of tile i.

template <int N, int K>
__device__ __forceinline__ void
gemm_load_tile(const __bf16* __restrict__ A, const __bf16* __restrict__ B,
               __bf16* dA, __bf16* dB, int m0, int n0, int k0, int tid) {
#pragma unroll
  for (int it = 0; it < 2; ++it) {     // A: 128x32 = 4096 bf16
    const int e  = (tid + it * 256) * 8;
    const int ra = e >> 5, ca = e & 31;
    copy16_g2l(&dA[e], &A[(size_t)(m0 + ra) * K + k0 + ca]);
  }
#pragma unroll
  for (int it = 0; it < 4; ++it) {     // B: 32x256 = 8192 bf16
    const int e  = (tid + it * 256) * 8;
    const int rb = e >> 8, cb = e & 255;
    copy16_g2l(&dB[e], &B[(size_t)(k0 + rb) * N + n0 + cb]);
  }
}

template <int N, int K>
__global__ void __launch_bounds__(256)
gemm_bf16_f32(const __bf16* __restrict__ A, const __bf16* __restrict__ B,
              float* __restrict__ C) {
  __shared__ __align__(16) __bf16 sA[2][128 * 32];
  __shared__ __align__(16) __bf16 sB[2][32 * 256];

  const int tid  = threadIdx.x;
  const int wave = tid >> 5;
  const int wm   = wave >> 2;  // 0..1  (64-row slab)
  const int wn   = wave & 3;   // 0..3  (64-col slab)
  const int m0   = blockIdx.x * 128;
  const int n0   = blockIdx.y * 256;

  v8f acc[4][4];
#pragma unroll
  for (int i = 0; i < 4; ++i)
#pragma unroll
    for (int j = 0; j < 4; ++j)
#pragma unroll
      for (int v = 0; v < 8; ++v) acc[i][j][v] = 0.f;

  gemm_load_tile<N, K>(A, B, sA[0], sB[0], m0, n0, 0, tid);

  for (int k0 = 0; k0 < K; k0 += 32) {
    const int buf = (k0 >> 5) & 1;
    if (k0 + 32 < K) {
      gemm_load_tile<N, K>(A, B, sA[buf ^ 1], sB[buf ^ 1], m0, n0, k0 + 32, tid);
      if (k0 + 64 < K) {  // L2 prefetch two tiles ahead
        const int e = tid * 8;
        __builtin_prefetch(&A[(size_t)(m0 + (e >> 5)) * K + k0 + 64 + (e & 31)], 0, 1);
        __builtin_prefetch(&B[(size_t)(k0 + 64 + (e >> 8)) * N + n0 + (e & 255)], 0, 1);
      }
      async_wait<6>();   // drain current tile's 6 loads; next tile's keep flying
    } else {
      async_wait<0>();
    }
    __syncthreads();

    v16bf af[4], bfc[4];
#pragma unroll
    for (int i = 0; i < 4; ++i)
      af[i] = load_A_frag(&sA[buf][(wm * 64 + i * 16) * 32], 32);
#pragma unroll
    for (int j = 0; j < 4; ++j)
      bfc[j] = load_B_frag(&sB[buf][wn * 64 + j * 16], 256);
#pragma unroll
    for (int i = 0; i < 4; ++i)
#pragma unroll
      for (int j = 0; j < 4; ++j)
        acc[i][j] = wmma_bf16(af[i], bfc[j], acc[i][j]);
    __syncthreads();
  }

  const int lane = tid & 31;
  const int nl   = lane & 15;
  const int mhh  = (lane >> 4) << 3;
#pragma unroll
  for (int i = 0; i < 4; ++i)
#pragma unroll
    for (int j = 0; j < 4; ++j) {
      float* cp = &C[(size_t)(m0 + wm * 64 + i * 16 + mhh) * N +
                     n0 + wn * 64 + j * 16 + nl];
#pragma unroll
      for (int v = 0; v < 8; ++v) cp[(size_t)v * N] = acc[i][j][v];
    }
}

// -------------------------- RoPE + layout pack -------------------------------
// proj: (B, S, nheads, 128) f32  ->  out: (B, nheads, S, 128) bf16 with RoPE.

__global__ void __launch_bounds__(256)
rope_pack(const float* __restrict__ proj, const float* __restrict__ cosb,
          const float* __restrict__ sinb, __bf16* __restrict__ out, int nheads) {
  long i = (long)blockIdx.x * 256 + threadIdx.x;
  const long total = (long)2 * nheads * 2048 * 64;
  if (i >= total) return;
  const int d = (int)(i & 63);
  long t = i >> 6;
  const int s = (int)(t % 2048); t /= 2048;
  const int h = (int)(t % nheads);
  const int b = (int)(t / nheads);
  const float* src = proj + ((size_t)(b * 2048 + s) * nheads + h) * 128;
  const float x1 = src[d], x2 = src[d + 64];
  const float c1 = cosb[s * 128 + d],      s1 = sinb[s * 128 + d];
  const float c2 = cosb[s * 128 + d + 64], s2 = sinb[s * 128 + d + 64];
  __bf16* dst = out + ((size_t)(b * nheads + h) * 2048 + s) * 128;
  dst[d]      = (__bf16)(x1 * c1 - x2 * s1);
  dst[d + 64] = (__bf16)(x2 * c2 + x1 * s2);
}

__global__ void __launch_bounds__(256)
pack_v(const float* __restrict__ proj, __bf16* __restrict__ out) {
  // (B, S, 8, 128) f32 -> (B, 8, S, 128) bf16
  long i = (long)blockIdx.x * 256 + threadIdx.x;
  const long total = (long)2 * 8 * 2048 * 128;
  if (i >= total) return;
  const int d = (int)(i & 127);
  long t = i >> 7;
  const int s = (int)(t % 2048); t /= 2048;
  const int h = (int)(t % 8);
  const int b = (int)(t / 8);
  out[((size_t)(b * 8 + h) * 2048 + s) * 128 + d] =
      (__bf16)proj[((size_t)(b * 2048 + s) * 8 + h) * 128 + d];
}

// ------------------------- flash attention (causal GQA) ----------------------
// grid (S/128, H, B); block 256 = 8 waves; wave w owns query rows
// [blk*128 + 16w, +16). Double-buffered K/V tiles (32 keys) via async loads.

__device__ __forceinline__ void
attn_load_kv(const __bf16* kh, const __bf16* vh, __bf16* dK, __bf16* dV,
             int kt, int tid) {
#pragma unroll
  for (int it = 0; it < 2; ++it) {
    const int e = (tid + it * 256) * 8;
    const int r = e >> 7, c = e & 127;
    copy16_g2l(&dK[e], &kh[(size_t)(kt * 32 + r) * 128 + c]);
    copy16_g2l(&dV[e], &vh[(size_t)(kt * 32 + r) * 128 + c]);
  }
}

__global__ void __launch_bounds__(256)
gqa_attn(const __bf16* __restrict__ Q, const __bf16* __restrict__ Kc,
         const __bf16* __restrict__ Vc, __bf16* __restrict__ ctx) {
  __shared__ __align__(16) __bf16 sK[2][32 * 128];
  __shared__ __align__(16) __bf16 sV[2][32 * 128];
  __shared__ __align__(4)  __bf16 sP[8][16 * 32];

  const int tid  = threadIdx.x;
  const int wave = tid >> 5;
  const int lane = tid & 31;
  const int h   = blockIdx.y;
  const int b   = blockIdx.z;
  const int kvh = h >> 2;  // GROUP = 4

  const __bf16* qh = Q  + ((size_t)(b * 32 + h)   * 2048) * 128;
  const __bf16* kh = Kc + ((size_t)(b * 8 + kvh) * 2048) * 128;
  const __bf16* vh = Vc + ((size_t)(b * 8 + kvh) * 2048) * 128;

  const int qBase = blockIdx.x * 128 + wave * 16;

  // Q fragments: 4 chunks of K=32 covering d=128 (A layout, straight from HBM)
  v16bf qf[4];
  {
    const int row = lane & 15;
    const int kg  = (lane >> 4) << 4;
#pragma unroll
    for (int c = 0; c < 4; ++c) {
#pragma unroll
      for (int v = 0; v < 8; ++v) {
        const int k = c * 32 + (v << 1) + kg;
        qf[c][2 * v]     = qh[(size_t)(qBase + row) * 128 + k];
        qf[c][2 * v + 1] = qh[(size_t)(qBase + row) * 128 + k + 1];
      }
    }
  }

  v8f o[8];
#pragma unroll
  for (int j = 0; j < 8; ++j)
#pragma unroll
    for (int v = 0; v < 8; ++v) o[j][v] = 0.f;
  float rm[8], rl[8];
#pragma unroll
  for (int v = 0; v < 8; ++v) { rm[v] = -1e30f; rl[v] = 0.f; }

  const int nl  = lane & 15;
  const int mhh = (lane >> 4) << 3;
  const float scale = 0.08838834764831845f;  // 1/sqrt(128)

  const int nkt = blockIdx.x * 4 + 4;  // causal: only key tiles <= block end

  attn_load_kv(kh, vh, sK[0], sV[0], 0, tid);

  for (int kt = 0; kt < nkt; ++kt) {
    const int buf = kt & 1;
    if (kt + 1 < nkt) {
      attn_load_kv(kh, vh, sK[buf ^ 1], sV[buf ^ 1], kt + 1, tid);
      if (kt + 2 < nkt) {
        const int e = tid * 8;
        __builtin_prefetch(&kh[(size_t)((kt + 2) * 32 + (e >> 7)) * 128 + (e & 127)], 0, 1);
        __builtin_prefetch(&vh[(size_t)((kt + 2) * 32 + (e >> 7)) * 128 + (e & 127)], 0, 1);
      }
      async_wait<4>();   // drain current tile's 4 loads; next tile's keep flying
    } else {
      async_wait<0>();
    }
    __syncthreads();

    // scores: (16 rows) x (32 keys) as two 16x16 accumulators
    v8f sc[2];
#pragma unroll
    for (int st = 0; st < 2; ++st) {
#pragma unroll
      for (int v = 0; v < 8; ++v) sc[st][v] = 0.f;
#pragma unroll
      for (int c = 0; c < 4; ++c) {
        v16bf kb = load_B_fragT(&sK[buf][(st * 16) * 128 + c * 32], 128);
        sc[st] = wmma_bf16(qf[c], kb, sc[st]);
      }
    }

    // online softmax (row stats per-VGPR: lanes<16 rows 0-7, lanes>=16 rows 8-15)
#pragma unroll
    for (int v = 0; v < 8; ++v) {
      const int row = qBase + mhh + v;
      const int kg0 = kt * 32 + nl;
      float s0 = sc[0][v] * scale;
      float s1 = sc[1][v] * scale;
      if (kg0      > row) s0 = -3.0e38f;
      if (kg0 + 16 > row) s1 = -3.0e38f;
      float tm = fmaxf(s0, s1);
#pragma unroll
      for (int off = 1; off < 16; off <<= 1)
        tm = fmaxf(tm, __shfl_xor(tm, off, 32));
      const float mnew  = fmaxf(rm[v], tm);
      const float alpha = __expf(rm[v] - mnew);
      rm[v] = mnew;
      const float p0 = __expf(s0 - mnew);
      const float p1 = __expf(s1 - mnew);
      float rs = p0 + p1;
#pragma unroll
      for (int off = 1; off < 16; off <<= 1)
        rs += __shfl_xor(rs, off, 32);
      rl[v] = rl[v] * alpha + rs;
#pragma unroll
      for (int j = 0; j < 8; ++j) o[j][v] *= alpha;
      // stage P (C-layout -> row-major) in per-wave LDS; same-wave DS ops are
      // in-order so the A-fragment reload below sees them.
      sP[wave][(mhh + v) * 32 + nl]      = (__bf16)p0;
      sP[wave][(mhh + v) * 32 + 16 + nl] = (__bf16)p1;
    }

    // O(16x128) += P(16x32) * V(32x128)
    v16bf pa = load_A_frag(&sP[wave][0], 32);
#pragma unroll
    for (int j = 0; j < 8; ++j) {
      v16bf vb = load_B_frag(&sV[buf][j * 16], 128);
      o[j] = wmma_bf16(pa, vb, o[j]);
    }
    __syncthreads();
  }

  // normalize and store ctx in (B, S, H*128) bf16 for the output projection
#pragma unroll
  for (int v = 0; v < 8; ++v) {
    const float inv = 1.f / rl[v];
    const int srow  = qBase + mhh + v;
#pragma unroll
    for (int j = 0; j < 8; ++j) {
      ctx[((size_t)(b * 2048 + srow)) * 4096 + h * 128 + j * 16 + nl] =
          (__bf16)(o[j][v] * inv);
    }
  }
}

// ------------------------------- launcher ------------------------------------

extern "C" void kernel_launch(void* const* d_in, const int* in_sizes, int n_in,
                              void* d_out, int out_size, void* d_ws,
                              size_t ws_size, hipStream_t stream) {
  (void)in_sizes; (void)n_in; (void)out_size; (void)ws_size;
  const float* x    = (const float*)d_in[0];
  const float* Wq   = (const float*)d_in[1];
  const float* Wk   = (const float*)d_in[2];
  const float* Wv   = (const float*)d_in[3];
  const float* Wo   = (const float*)d_in[4];
  const float* cosb = (const float*)d_in[5];
  const float* sinb = (const float*)d_in[6];

  const long MB = 4096;             // B*S rows
  const long K  = 4096;             // D_IN
  const long NQ = 4096, NKV = 1024;

  char* ws = (char*)d_ws;
  size_t off = 0;
  auto alloc = [&](size_t bytes) {
    void* p = ws + off;
    off += (bytes + 255) & ~(size_t)255;
    return p;
  };

  __bf16* x_bf  = (__bf16*)alloc(MB * K * 2);
  __bf16* Wq_bf = (__bf16*)alloc(K * NQ * 2);
  __bf16* Wk_bf = (__bf16*)alloc(K * NKV * 2);
  __bf16* Wv_bf = (__bf16*)alloc(K * NKV * 2);
  __bf16* Wo_bf = (__bf16*)alloc(NQ * NQ * 2);
  float*  qp    = (float*) alloc(MB * NQ * 4);
  float*  kp    = (float*) alloc(MB * NKV * 4);
  float*  vp    = (float*) alloc(MB * NKV * 4);
  __bf16* Qb    = (__bf16*)alloc(MB * NQ * 2);
  __bf16* Kb    = (__bf16*)alloc(MB * NKV * 2);
  __bf16* Vb    = (__bf16*)alloc(MB * NKV * 2);
  __bf16* ctx   = (__bf16*)alloc(MB * NQ * 2);

  auto cvt = [&](const float* src, __bf16* dst, long n) {
    f32_to_bf16<<<dim3((unsigned)((n + 255) / 256)), 256, 0, stream>>>(src, dst, n);
  };
  cvt(x,  x_bf,  MB * K);
  cvt(Wq, Wq_bf, K * NQ);
  cvt(Wk, Wk_bf, K * NKV);
  cvt(Wv, Wv_bf, K * NKV);
  cvt(Wo, Wo_bf, NQ * NQ);

  gemm_bf16_f32<4096, 4096><<<dim3(32, 16), 256, 0, stream>>>(x_bf, Wq_bf, qp);
  gemm_bf16_f32<1024, 4096><<<dim3(32, 4),  256, 0, stream>>>(x_bf, Wk_bf, kp);
  gemm_bf16_f32<1024, 4096><<<dim3(32, 4),  256, 0, stream>>>(x_bf, Wv_bf, vp);

  {
    long n = 2L * 32 * 2048 * 64;
    rope_pack<<<dim3((unsigned)((n + 255) / 256)), 256, 0, stream>>>(qp, cosb, sinb, Qb, 32);
    n = 2L * 8 * 2048 * 64;
    rope_pack<<<dim3((unsigned)((n + 255) / 256)), 256, 0, stream>>>(kp, cosb, sinb, Kb, 8);
    n = 2L * 8 * 2048 * 128;
    pack_v<<<dim3((unsigned)((n + 255) / 256)), 256, 0, stream>>>(vp, Vb);
  }

  gqa_attn<<<dim3(16, 32, 2), 256, 0, stream>>>(Qb, Kb, Vb, ctx);

  gemm_bf16_f32<4096, 4096><<<dim3(32, 16), 256, 0, stream>>>(ctx, Wo_bf,
                                                              (float*)d_out);
}
